// AttentionAugmentedNSDE_28286654611811
// MI455X (gfx1250) — matrix operand order repaired and checked by reading, not compiled, not run on hardware
//
#include <hip/hip_runtime.h>

// ---------------------------------------------------------------------------
// AttentionAugmentedNSDE for MI455X (gfx1250, wave32, WMMA 16x16x32 f16)
// Register-blocked GEMM (32x64 per wave), LDS-shared weight panels staged
// with async global->LDS loads (b128 payload = int4 in AS1 -> AS3).
// ---------------------------------------------------------------------------

typedef __attribute__((ext_vector_type(16))) _Float16 v16h;
typedef __attribute__((ext_vector_type(8)))  _Float16 v8h;
typedef __attribute__((ext_vector_type(8)))  float    v8f;
typedef __attribute__((ext_vector_type(4)))  int      v4i;

#define BDIM 64
#define TDIM 512
#define DDIM 128
#define HDIM 128
#define LDIM 256
#define MROWS (BDIM * TDIM)   // 32768 fused rows

#if defined(__has_builtin)
#if __has_builtin(__builtin_amdgcn_global_load_async_to_lds_b128) && \
    __has_builtin(__builtin_amdgcn_s_wait_asynccnt)
#define USE_ASYNC_LDS 1
#endif
#endif

#define GLOBAL_AS __attribute__((address_space(1)))
#define SHARED_AS __attribute__((address_space(3)))

// D = A(16x32) * B(32x16) + C, f32 accumulate
__device__ __forceinline__ v8f wmma_f16(v16h a, v16h b, v8f c) {
  return __builtin_amdgcn_wmma_f32_16x16x32_f16(
      /*neg_a=*/false, a, /*neg_b=*/false, b,
      /*c_mod=*/(short)0, c, /*reuse_a=*/false, /*reuse_b=*/false);
}

// A fragment: 16x32 f16 tile, row-major source, row = lane&15,
// lanes 0-15 hold K {0..7,16..23}+k0, lanes 16-31 hold {8..15,24..31}+k0
__device__ __forceinline__ v16h load_a_g(const _Float16* base, int lda, int lane, int k0) {
  const int r  = lane & 15;
  const int kb = k0 + ((lane >> 4) << 3);
  const _Float16* p = base + (size_t)r * lda + kb;
  v8h lo = *reinterpret_cast<const v8h*>(p);
  v8h hi = *reinterpret_cast<const v8h*>(p + 16);
  v16h o;
#pragma unroll
  for (int i = 0; i < 8; ++i) { o[i] = lo[i]; o[i + 8] = hi[i]; }
  return o;
}

// B fragment: 32x16 f16 tile from W^T layout [col][K]; col = lane&15,
// lanes 0-15 hold K {0..7,16..23}+k0, lanes 16-31 hold {8..15,24..31}+k0
__device__ __forceinline__ v16h load_b_g(const _Float16* tile, int ldb, int lane, int k0) {
  const int c  = lane & 15;
  const int kb = k0 + ((lane >> 4) << 3);
  const _Float16* p = tile + (size_t)c * ldb + kb;
  v8h lo = *reinterpret_cast<const v8h*>(p);
  v8h hi = *reinterpret_cast<const v8h*>(p + 16);
  v16h o;
#pragma unroll
  for (int i = 0; i < 8; ++i) { o[i] = lo[i]; o[i + 8] = hi[i]; }
  return o;
}

// ---------------------------------------------------------------------------
// elementwise f32 -> f16
__global__ void f32_to_f16_kernel(const float* __restrict__ in,
                                  _Float16* __restrict__ out, int n) {
  for (int i = blockIdx.x * blockDim.x + threadIdx.x; i < n;
       i += gridDim.x * blockDim.x)
    out[i] = (_Float16)in[i];
}

// W[K][N] f32  ->  Wt[N][K] f16   (K, N multiples of 32)
__global__ __launch_bounds__(256) void transpose_cvt_kernel(
    const float* __restrict__ W, _Float16* __restrict__ Wt, int K, int N) {
  __shared__ float tile[32][33];
  const int k0 = blockIdx.x * 32, n0 = blockIdx.y * 32;
  const int tx = threadIdx.x & 31, ty = threadIdx.x >> 5;
  for (int i = ty; i < 32; i += 8)
    tile[i][tx] = W[(size_t)(k0 + i) * N + (n0 + tx)];
  __syncthreads();
  for (int i = ty; i < 32; i += 8)
    Wt[(size_t)(n0 + i) * K + (k0 + tx)] = (_Float16)tile[tx][i];
}

// V[b][s][d] f16 -> Vt[b][d][s] f16
__global__ __launch_bounds__(256) void transpose_v_kernel(
    const _Float16* __restrict__ V, _Float16* __restrict__ Vt) {
  __shared__ _Float16 tile[32][33];
  const int b = blockIdx.z;
  const int s0 = blockIdx.x * 32, d0 = blockIdx.y * 32;
  const int tx = threadIdx.x & 31, ty = threadIdx.x >> 5;
  const _Float16* Vb = V + (size_t)b * TDIM * DDIM;
  _Float16* Vtb = Vt + (size_t)b * DDIM * TDIM;
  for (int i = ty; i < 32; i += 8)
    tile[i][tx] = Vb[(size_t)(s0 + i) * DDIM + (d0 + tx)];
  __syncthreads();
  for (int i = ty; i < 32; i += 8)
    Vtb[(size_t)(d0 + i) * TDIM + (s0 + tx)] = tile[tx][i];
}

// ---------------------------------------------------------------------------
// Register-blocked GEMM: Y = act(X[M,K] @ W(^T stored as [N][K]) + bias)
// Block = 8 waves = 256 rows x 64 cols. The 64xK weight panel is staged into
// LDS once per block (async global->LDS when available) and shared by all
// waves; each wave computes a 32x64 register block (2x4 WMMA tiles).
// Requires M % 256 == 0, N % 64 == 0, K <= 256. act: 0=none, 1=relu
__global__ __launch_bounds__(256) void gemm_f16_kernel(
    const _Float16* __restrict__ X, int lda,
    const _Float16* __restrict__ Wt,
    const float* __restrict__ bias,
    _Float16* __restrict__ Yh, float* __restrict__ Yf, int ldy,
    int M, int N, int K, int act) {
  __shared__ _Float16 Bs[64][264];   // 64 cols x (K<=256 + 8 pad): 33 KB

  const int gn_cnt = N >> 6;
  const int bm = (blockIdx.x / gn_cnt) << 8;     // 256 rows per block
  const int bn = (blockIdx.x % gn_cnt) << 6;     // 64 cols per block
  const int tid = threadIdx.x, lane = tid & 31, wave = tid >> 5;

  // --- stage 64xK weight panel into LDS (16B vectors, coalesced) ---
  {
    const int kv = K >> 3;                       // 16B vectors per row
    for (int v = tid; v < 64 * kv; v += 256) {
      const int c  = v / kv;
      const int kk = (v - c * kv) << 3;
      const _Float16* gp = Wt + (size_t)(bn + c) * K + kk;
#ifdef USE_ASYNC_LDS
      __builtin_amdgcn_global_load_async_to_lds_b128(
          (GLOBAL_AS v4i*)gp, (SHARED_AS v4i*)&Bs[c][kk], 0, 0);
#else
      *reinterpret_cast<v8h*>(&Bs[c][kk]) = *reinterpret_cast<const v8h*>(gp);
#endif
    }
  }
#ifdef USE_ASYNC_LDS
  __builtin_amdgcn_s_wait_asynccnt(0);
#endif
  __syncthreads();

  // --- main loop: wave owns rows [bm + wave*32, +32), cols [bn, bn+64) ---
  const int rw = bm + wave * 32;
  const _Float16* Xr = X + (size_t)rw * lda;
  v8f acc[2][4] = {};
  for (int k0 = 0; k0 < K; k0 += 32) {
    v16h a0 = load_a_g(Xr, lda, lane, k0);
    v16h a1 = load_a_g(Xr + (size_t)16 * lda, lda, lane, k0);
    const int c  = lane & 15;
    const int kb = k0 + ((lane >> 4) << 3);
#pragma unroll
    for (int j = 0; j < 4; ++j) {
      v8h lo = *reinterpret_cast<const v8h*>(&Bs[j * 16 + c][kb]);     // ds_load
      v8h hi = *reinterpret_cast<const v8h*>(&Bs[j * 16 + c][kb + 16]);
      v16h bv;
#pragma unroll
      for (int i = 0; i < 8; ++i) { bv[i] = lo[i]; bv[i + 8] = hi[i]; }
      acc[0][j] = wmma_f16(a0, bv, acc[0][j]);
      acc[1][j] = wmma_f16(a1, bv, acc[1][j]);
    }
  }

  // --- epilogue ---
  const int rb = (lane >> 4) << 3;
#pragma unroll
  for (int i = 0; i < 2; ++i)
#pragma unroll
    for (int j = 0; j < 4; ++j) {
      const int col = bn + j * 16 + (lane & 15);
      const float bz = bias ? bias[col] : 0.f;
#pragma unroll
      for (int r = 0; r < 8; ++r) {
        float v = acc[i][j][r] + bz;
        if (act == 1) v = fmaxf(v, 0.f);
        const size_t idx = (size_t)(rw + i * 16 + rb + r) * ldy + col;
        if (Yh) Yh[idx] = (_Float16)v;
        else    Yf[idx] = v;
      }
    }
}

// ---------------------------------------------------------------------------
// Attention: one block per (batch, 16-query tile). 8 waves.
// S = Q K^T / sqrt(D) -> softmax -> context = P V, written into fusion[:,0:128]
__global__ __launch_bounds__(256) void attention_kernel(
    const _Float16* __restrict__ Q, const _Float16* __restrict__ Km,
    const _Float16* __restrict__ Vt, _Float16* __restrict__ fusion) {
  __shared__ float    S[16][TDIM];     // 32 KB
  __shared__ _Float16 P[16][TDIM];     // 16 KB
  __shared__ float    red[16][16];
  __shared__ float    rmax[16];
  __shared__ float    rsum[16];

  const int b = blockIdx.y, t0 = blockIdx.x * 16;
  const int lane = threadIdx.x & 31, wave = threadIdx.x >> 5;
  const _Float16* Qb = Q  + ((size_t)b * TDIM + t0) * DDIM;
  const _Float16* Kb = Km + (size_t)b * TDIM * DDIM;   // [s][d] == B[col=s][k=d]
  const _Float16* Vb = Vt + (size_t)b * DDIM * TDIM;   // [d][s] == B[col=d][k=s]

  // Q tile A-fragments (K = 128 -> 4 fragments), reused across all 32 col tiles
  v16h afrag[4];
#pragma unroll
  for (int kk = 0; kk < 4; ++kk) afrag[kk] = load_a_g(Qb, DDIM, lane, kk * 32);

  const float scale = 0.088388347648318447f;  // 1/sqrt(128)
  for (int ct = wave; ct < TDIM / 16; ct += 8) {
    v8f acc = {};
#pragma unroll
    for (int kk = 0; kk < 4; ++kk) {
      v16h bv = load_b_g(Kb + (size_t)ct * 16 * DDIM, DDIM, lane, kk * 32);
      acc = wmma_f16(afrag[kk], bv, acc);
    }
    const int col = ct * 16 + (lane & 15);
    const int rb  = (lane >> 4) << 3;
#pragma unroll
    for (int r = 0; r < 8; ++r) S[rb + r][col] = acc[r] * scale;
  }
  __syncthreads();

  // row-wise softmax: 16 threads per row, 32 cols each
  const int row = threadIdx.x >> 4, seg = threadIdx.x & 15;
  float m = -3.0e38f;
  for (int c = seg * 32; c < seg * 32 + 32; ++c) m = fmaxf(m, S[row][c]);
  red[row][seg] = m;
  __syncthreads();
  if (seg == 0) {
    float mm = red[row][0];
#pragma unroll
    for (int i = 1; i < 16; ++i) mm = fmaxf(mm, red[row][i]);
    rmax[row] = mm;
  }
  __syncthreads();
  const float mm = rmax[row];
  float s = 0.f;
  for (int c = seg * 32; c < seg * 32 + 32; ++c) {
    float e = __expf(S[row][c] - mm);
    P[row][c] = (_Float16)e;
    s += e;
  }
  red[row][seg] = s;
  __syncthreads();
  if (seg == 0) {
    float ss = 0.f;
#pragma unroll
    for (int i = 0; i < 16; ++i) ss += red[row][i];
    rsum[row] = 1.0f / ss;
  }
  __syncthreads();

  // context tile: wave w owns output columns [w*16, w*16+16); K = 512
  v8f acc = {};
  for (int k0 = 0; k0 < TDIM; k0 += 32) {
    const int r  = lane & 15;
    const int kb = k0 + ((lane >> 4) << 3);
    v8h lo = *reinterpret_cast<const v8h*>(&P[r][kb]);        // ds_load_b128
    v8h hi = *reinterpret_cast<const v8h*>(&P[r][kb + 16]);
    v16h av;
#pragma unroll
    for (int i = 0; i < 8; ++i) { av[i] = lo[i]; av[i + 8] = hi[i]; }
    v16h bv = load_b_g(Vb + (size_t)wave * 16 * TDIM, TDIM, lane, k0);
    acc = wmma_f16(av, bv, acc);
  }
  const int col = wave * 16 + (lane & 15);
  const int rb  = (lane >> 4) << 3;
#pragma unroll
  for (int r = 0; r < 8; ++r) {
    const int rr = rb + r;
    fusion[((size_t)b * TDIM + t0 + rr) * (DDIM + HDIM) + col] =
        (_Float16)(acc[r] * rsum[rr]);
  }
}

// ---------------------------------------------------------------------------
// Sequential neural-SDE scan: single workgroup, 16 waves, LDS ping-pong.
// Each wave owns one 16-row tile and a 2-4 tile column strip, so one LDS
// A-fragment feeds 2-4 WMMAs per k-step.
struct SdeArgs {
  const float* noise;      // [T-1][B][H]
  _Float16*    fusion;     // [B][T][256], writes cols 128..255
  float*       yG;         // [B][H] f32 state
  _Float16*    ybG;        // [B][H] f16 state copy
  float*       DG;         // drift output
  float*       EG;         // diffusion output
  const _Float16* W[8];    // drift {in,fc0,fc1,out}, diff {in,fc0,fc1,out} (T)
  const float*    Bs[8];
};

__global__ __launch_bounds__(512) void sde_scan_kernel(SdeArgs a) {
  __shared__ _Float16 Hs[2][BDIM][LDIM];   // 64 KB ping-pong activations
  const int tid = threadIdx.x, lane = tid & 31, wave = tid >> 5;  // 16 waves

  // y0 = 0, hidden[:,0,:] = 0
  for (int i = tid; i < BDIM * HDIM; i += 512) {
    a.yG[i] = 0.f; a.ybG[i] = (_Float16)0.f;
    const int b = i >> 7, h = i & 127;
    a.fusion[((size_t)b * TDIM) * (DDIM + HDIM) + DDIM + h] = (_Float16)0.f;
  }
  __syncthreads();

  const float dt = 1.0f / (float)(TDIM - 1);
  const float sqdt = sqrtf(dt);

  // one linear layer: src LDS -> dst LDS (or f32 global for the out layer)
  auto layer = [&](int sb, int db, const _Float16* W, const float* Bb,
                   int N, int K, int act, float* gout) {
    const int C   = N >> 6;              // col tiles per wave: 4 (N=256), 2 (128)
    const int tm  = wave & 3;            // row tile (64 rows -> 4 tiles)
    const int tn0 = (wave >> 2) * C;     // column strip start
    __builtin_prefetch(W + (size_t)tn0 * 16 * K, 0, 0);   // L2-resident hint
    v8f acc[4] = {};
    for (int k0 = 0; k0 < K; k0 += 32) {
      const int r  = tm * 16 + (lane & 15);
      const int kb = k0 + ((lane >> 4) << 3);
      v8h lo = *reinterpret_cast<const v8h*>(&Hs[sb][r][kb]);
      v8h hi = *reinterpret_cast<const v8h*>(&Hs[sb][r][kb + 16]);
      v16h av;
#pragma unroll
      for (int i = 0; i < 8; ++i) { av[i] = lo[i]; av[i + 8] = hi[i]; }
#pragma unroll 4
      for (int j = 0; j < C; ++j) {
        v16h bv = load_b_g(W + (size_t)(tn0 + j) * 16 * K, K, lane, k0);
        acc[j] = wmma_f16(av, bv, acc[j]);
      }
    }
    const int rb = (lane >> 4) << 3;
    for (int j = 0; j < C; ++j) {
      const int col = (tn0 + j) * 16 + (lane & 15);
      const float bz = Bb[col];
#pragma unroll
      for (int r = 0; r < 8; ++r) {
        float v = acc[j][r] + bz;
        if (act == 1)      v = fmaxf(v, 0.f);
        else if (act == 2) v = 1.f / (1.f + __expf(-v));   // sigmoid
        const int row = tm * 16 + rb + r;
        if (gout) gout[row * HDIM + col] = v;              // N == HDIM here
        else      Hs[db][row][col] = (_Float16)v;
      }
    }
    __syncthreads();
  };

  for (int t = 0; t < TDIM - 1; ++t) {
    for (int net = 0; net < 2; ++net) {
      // stage y (f16) into Hs[1][:, 0:128]
      for (int i = tid; i < BDIM * HDIM; i += 512)
        Hs[1][i >> 7][i & 127] = a.ybG[i];
      __syncthreads();
      const _Float16* const* W  = &a.W[net * 4];
      const float* const*    Bb = &a.Bs[net * 4];
      float* gout = (net == 0) ? a.DG : a.EG;
      layer(1, 0, W[0], Bb[0], LDIM, HDIM, 0, nullptr);   // in      (no act)
      layer(0, 1, W[1], Bb[1], LDIM, LDIM, 1, nullptr);   // fc0 +relu
      layer(1, 0, W[2], Bb[2], LDIM, LDIM, 1, nullptr);   // fc1 +relu
      layer(0, 0, W[3], Bb[3], HDIM, LDIM, 2, gout);      // out +sigmoid
    }
    // Euler-Maruyama update + write hidden[:, t+1, :]
    const float* nz = a.noise + (size_t)t * BDIM * HDIM;
    for (int i = tid; i < BDIM * HDIM; i += 512) {
      const float yv = a.yG[i] + a.DG[i] * dt + a.EG[i] * (sqdt * nz[i]);
      a.yG[i] = yv;
      const _Float16 yb = (_Float16)yv;
      a.ybG[i] = yb;
      const int b = i >> 7, h = i & 127;
      a.fusion[((size_t)b * TDIM + (t + 1)) * (DDIM + HDIM) + DDIM + h] = yb;
    }
    __syncthreads();
  }
}

// ---------------------------------------------------------------------------
extern "C" void kernel_launch(void* const* d_in, const int* in_sizes, int n_in,
                              void* d_out, int out_size, void* d_ws, size_t ws_size,
                              hipStream_t stream) {
  (void)in_sizes; (void)n_in; (void)out_size; (void)ws_size;
  // Inputs flattened in setup_inputs() insertion order:
  // 0:x 1:noise | 2..7 attn q/k/v (w,b) | 8..15 drift in/fc0/fc1/out (w,b)
  // 16..23 diff | 24,25 fc_in | 26..29 fc_block | 30,31 fc_out
  const float* x     = (const float*)d_in[0];
  const float* noise = (const float*)d_in[1];

  char* base = (char*)d_ws;
  size_t off = 0;
  auto alloc = [&](size_t bytes) -> void* {
    off = (off + 255) & ~(size_t)255;
    void* p = base + off;
    off += bytes;
    return p;
  };

  const size_t NX = (size_t)MROWS * DDIM;               // 4,194,304
  _Float16* Xh = (_Float16*)alloc(NX * 2);
  _Float16* Qh = (_Float16*)alloc(NX * 2);
  _Float16* Kh = (_Float16*)alloc(NX * 2);
  _Float16* Fh = (_Float16*)alloc((size_t)MROWS * 256 * 2);  // [B,T,256] concat
  _Float16* H1 = (_Float16*)alloc((size_t)MROWS * 256 * 2);
  _Float16* H2 = (_Float16*)alloc((size_t)MROWS * 256 * 2);
  _Float16* Vtmp = H1;   // alias: V raw, dead before H1 is first written
  _Float16* Vth  = H2;   // alias: V^T,  dead before H2 is first written

  // transposed f16 weights
  _Float16* qT   = (_Float16*)alloc(16384 * 2);
  _Float16* kT   = (_Float16*)alloc(16384 * 2);
  _Float16* vT   = (_Float16*)alloc(16384 * 2);
  _Float16* WT[8];  // drift in/fc0/fc1/out, diff in/fc0/fc1/out
  const int wsz[4] = {128 * 256, 256 * 256, 256 * 256, 256 * 128};
  for (int n = 0; n < 2; ++n)
    for (int l = 0; l < 4; ++l)
      WT[n * 4 + l] = (_Float16*)alloc((size_t)wsz[l] * 2);
  _Float16* fcInT  = (_Float16*)alloc((size_t)256 * 256 * 2);
  _Float16* fcb0T  = (_Float16*)alloc((size_t)256 * 256 * 2);
  _Float16* fcb1T  = (_Float16*)alloc((size_t)256 * 256 * 2);
  _Float16* fcOutT = (_Float16*)alloc((size_t)256 * 128 * 2);
  float*    yG  = (float*)alloc((size_t)BDIM * HDIM * 4);
  _Float16* ybG = (_Float16*)alloc((size_t)BDIM * HDIM * 2);
  float*    DG  = (float*)alloc((size_t)BDIM * HDIM * 4);
  float*    EG  = (float*)alloc((size_t)BDIM * HDIM * 4);

  // 1) x -> f16
  f32_to_f16_kernel<<<4096, 256, 0, stream>>>(x, Xh, (int)NX);

  // 2) transpose+convert all weight matrices
  struct TW { const float* w; _Float16* t; int K, N; };
  TW tw[15] = {
    {(const float*)d_in[2],  qT,     128, 128},
    {(const float*)d_in[4],  kT,     128, 128},
    {(const float*)d_in[6],  vT,     128, 128},
    {(const float*)d_in[8],  WT[0],  128, 256},
    {(const float*)d_in[10], WT[1],  256, 256},
    {(const float*)d_in[12], WT[2],  256, 256},
    {(const float*)d_in[14], WT[3],  256, 128},
    {(const float*)d_in[16], WT[4],  128, 256},
    {(const float*)d_in[18], WT[5],  256, 256},
    {(const float*)d_in[20], WT[6],  256, 256},
    {(const float*)d_in[22], WT[7],  256, 128},
    {(const float*)d_in[24], fcInT,  256, 256},
    {(const float*)d_in[26], fcb0T,  256, 256},
    {(const float*)d_in[28], fcb1T,  256, 256},
    {(const float*)d_in[30], fcOutT, 256, 128},
  };
  for (int i = 0; i < 15; ++i)
    transpose_cvt_kernel<<<dim3(tw[i].K / 32, tw[i].N / 32), 256, 0, stream>>>(
        tw[i].w, tw[i].t, tw[i].K, tw[i].N);

  auto gemm = [&](const _Float16* X, int lda, const _Float16* Wt, const float* bias,
                  _Float16* Yh, float* Yf, int ldy, int M, int N, int K, int act) {
    const int blocks = (M >> 8) * (N >> 6);   // 256-row x 64-col blocks
    gemm_f16_kernel<<<blocks, 256, 0, stream>>>(X, lda, Wt, bias, Yh, Yf, ldy,
                                                M, N, K, act);
  };

  // 3) QKV projections (f16 WMMA, f32 accumulate)
  gemm(Xh, DDIM, qT, (const float*)d_in[3], Qh,   nullptr, DDIM, MROWS, DDIM, DDIM, 0);
  gemm(Xh, DDIM, kT, (const float*)d_in[5], Kh,   nullptr, DDIM, MROWS, DDIM, DDIM, 0);
  gemm(Xh, DDIM, vT, (const float*)d_in[7], Vtmp, nullptr, DDIM, MROWS, DDIM, DDIM, 0);
  transpose_v_kernel<<<dim3(TDIM / 32, DDIM / 32, BDIM), 256, 0, stream>>>(Vtmp, Vth);

  // 4) attention -> fusion[:, :, 0:128]
  attention_kernel<<<dim3(TDIM / 16, BDIM), 256, 0, stream>>>(Qh, Kh, Vth, Fh);

  // 5) sequential SDE scan -> fusion[:, :, 128:256]
  SdeArgs sa;
  sa.noise = noise; sa.fusion = Fh;
  sa.yG = yG; sa.ybG = ybG; sa.DG = DG; sa.EG = EG;
  const int bidx[8] = {9, 11, 13, 15, 17, 19, 21, 23};
  for (int i = 0; i < 8; ++i) { sa.W[i] = WT[i]; sa.Bs[i] = (const float*)d_in[bidx[i]]; }
  sde_scan_kernel<<<1, 512, 0, stream>>>(sa);

  // 6) fusion MLP -> d_out (f32)
  gemm(Fh, 256, fcInT, (const float*)d_in[25], H1, nullptr, 256, MROWS, 256, 256, 0);
  gemm(H1, 256, fcb0T, (const float*)d_in[27], H2, nullptr, 256, MROWS, 256, 256, 1);
  gemm(H2, 256, fcb1T, (const float*)d_in[29], H1, nullptr, 256, MROWS, 256, 256, 1);
  gemm(H1, 256, fcOutT, (const float*)d_in[31], nullptr, (float*)d_out, DDIM,
       MROWS, DDIM, 256, 0);
}
